// MoE_28166395527683
// MI455X (gfx1250) — matrix operand order repaired and checked by reading, not compiled, not run on hardware
//
#include <hip/hip_runtime.h>
#include <cstdint>

// ---------------- problem constants ----------------
#define M_TOK   16384
#define K_DIM   1024
#define E_NUM   8
#define N_DIM   5632
#define HALF_N  2816
#define TOPK    2
#define CAP     (M_TOK * TOPK)        // worst-case rows per expert
#define ROWS    128                   // gathered token rows per block
#define A_STR   (K_DIM + 8)           // padded LDS row stride (elems)
#define HC      1024                  // phase-3 LDS h2 chunk columns
#define C_STR   (HC + 8)              // chunk LDS stride (elems) == A_STR
#define H2_PAD  3072                  // h2 workspace row stride (elems), zero-padded past 2816
#define HS_STR  (128 + 8)             // h2 LDS staging stride (elems)

#define LDS_A_BYTES   (ROWS * A_STR * 2)       // 264192
#define LDS_HS_BYTES  (ROWS * HS_STR * 2)      // 34816
#define LDS_TOTAL     (LDS_A_BYTES + LDS_HS_BYTES + ROWS * 8)

typedef _Float16 f16;
typedef __attribute__((ext_vector_type(16))) _Float16 v16h;
typedef __attribute__((ext_vector_type(8)))  _Float16 v8h;
typedef __attribute__((ext_vector_type(8)))  float    v8f;
typedef __attribute__((ext_vector_type(4)))  float    v4f;

// ---------------- WMMA + fragment helpers ----------------
__device__ __forceinline__ v8f wmma_f16(v16h a, v16h b, v8f c) {
  return __builtin_amdgcn_wmma_f32_16x16x32_f16(false, a, false, b, (short)0, c, false, false);
}

// A-fragment (16x32, f16) from LDS row-major tile.
// lanes 0-15: row=lane, K {0..7,16..23}; lanes 16-31: row=lane-16, K {8..15,24..31}
__device__ __forceinline__ v16h load_a_lds(const char* base, int stride_b, int row0, int k0) {
  int lane = threadIdx.x & 31;
  int half = lane >> 4;
  const char* p = base + (row0 + (lane & 15)) * stride_b + k0 * 2 + half * 16;
  union { v16h v; v4f f[2]; } u;
  u.f[0] = *(const v4f*)(p);
  u.f[1] = *(const v4f*)(p + 32);
  return u.v;
}

// A-fragment from global, operand row-major (row = M, K contiguous)
__device__ __forceinline__ v16h load_a_glb(const f16* rowbase, long rstride, int row0, int k0) {
  int lane = threadIdx.x & 31;
  int half = lane >> 4;
  const f16* p = rowbase + (long)(row0 + (lane & 15)) * rstride + k0 + half * 8;
  union { v16h v; v4f f[2]; } u;
  u.f[0] = *(const v4f*)(p);
  u.f[1] = *(const v4f*)(p + 16);
  return u.v;
}

// B-fragment (32x16) from global, operand stored transposed as N x K (K contiguous).
// n = lane%16; lanes 0-15 hold K {0..7,16..23}, lanes 16-31 hold K {8..15,24..31}
__device__ __forceinline__ v16h load_b_glb(const f16* rowbase, long rstride, int n0, int k0) {
  int lane = threadIdx.x & 31;
  int half = lane >> 4;
  const f16* p = rowbase + (long)(n0 + (lane & 15)) * rstride + k0 + half * 8;
  union { v16h v; v4f f[2]; } u;
  u.f[0] = *(const v4f*)(p);
  u.f[1] = *(const v4f*)(p + 16);
  return u.v;
}

// B-fragment from LDS (same N x K transposed layout)
__device__ __forceinline__ v16h load_b_lds(const char* base, int stride_b, int n0, int k0) {
  int lane = threadIdx.x & 31;
  int half = lane >> 4;
  const char* p = base + (n0 + (lane & 15)) * stride_b + k0 * 2 + half * 16;
  union { v16h v; v4f f[2]; } u;
  u.f[0] = *(const v4f*)(p);
  u.f[1] = *(const v4f*)(p + 32);
  return u.v;
}

// gfx1250 async global->LDS copy (16B per lane), tracked by ASYNCcnt
__device__ __forceinline__ void async_ld_b128(unsigned lds_addr, const void* gptr) {
  asm volatile("global_load_async_to_lds_b128 %0, %1, off"
               :: "v"(lds_addr), "v"((unsigned long long)(uintptr_t)gptr)
               : "memory");
}
__device__ __forceinline__ void async_wait0() {
  asm volatile("s_wait_asynccnt 0x0" ::: "memory");
}

// fast silu: g * rcp(1 + exp(-g)); v_rcp_f32 (~1 ulp) << f16 storage precision
__device__ __forceinline__ float silu_fast(float g) {
  return g * __builtin_amdgcn_rcpf(1.f + __expf(-g));
}

// ---------------- small utility kernels ----------------
__global__ void zero_f32_kernel(float* p, long n) {
  long i = ((long)blockIdx.x * blockDim.x + threadIdx.x) * 4;
  long stride = (long)gridDim.x * blockDim.x * 4;
  for (; i < n; i += stride) {
    v4f z = {};
    *(v4f*)(p + i) = z;
  }
}

__global__ void cvt_f32_f16_kernel(const float* __restrict__ src, f16* __restrict__ dst, long n) {
  long i = ((long)blockIdx.x * blockDim.x + threadIdx.x) * 8;
  long stride = (long)gridDim.x * blockDim.x * 8;
  for (; i < n; i += stride) {
    v4f a = *(const v4f*)(src + i);
    v4f b = *(const v4f*)(src + i + 4);
    v8h o;
    o[0] = (f16)a[0]; o[1] = (f16)a[1]; o[2] = (f16)a[2]; o[3] = (f16)a[3];
    o[4] = (f16)b[0]; o[5] = (f16)b[1]; o[6] = (f16)b[2]; o[7] = (f16)b[3];
    *(v8h*)(dst + i) = o;
  }
}

__global__ void route_scatter_kernel(const int* __restrict__ ids,
                                     int* __restrict__ counts,
                                     int* __restrict__ list) {
  int i = blockIdx.x * blockDim.x + threadIdx.x;
  if (i >= M_TOK * TOPK) return;
  int e = ids[i];
  if (e < 0 || e >= E_NUM) return;
  int pos = atomicAdd(&counts[e], 1);
  list[e * CAP + pos] = i;
}

__global__ void prefix_kernel(const int* __restrict__ counts, int* __restrict__ prefix) {
  if (threadIdx.x == 0) {
    int run = 0;
    for (int e = 0; e < E_NUM; ++e) { prefix[e] = run; run += counts[e]; }
  }
}

// ---------------- fused MoE expert kernel ----------------
// grid: (CAP/ROWS, E), block: 256 threads (8 waves)
__global__ void __launch_bounds__(256)
moe_fused_kernel(const f16* __restrict__ hid,
                 const f16* __restrict__ w1,
                 const f16* __restrict__ w2,
                 const float* __restrict__ topw,
                 const int* __restrict__ counts,
                 const int* __restrict__ prefix,
                 const int* __restrict__ list,
                 f16* __restrict__ h2ws,
                 float* __restrict__ out) {
  extern __shared__ char smem[];
  char*  Atile  = smem;                       // ROWS x A_STR f16  (reused as h2 chunk in phase 3)
  char*  Hstage = smem + LDS_A_BYTES;         // ROWS x HS_STR f16
  int*   rowM   = (int*)(smem + LDS_A_BYTES + LDS_HS_BYTES);
  float* rowW   = (float*)(rowM + ROWS);

  const int e    = blockIdx.y;
  const int cnt  = counts[e];
  const int base = blockIdx.x * ROWS;
  if (base >= cnt) return;

  const int tid   = threadIdx.x;
  const int lane  = tid & 31;
  const int wave  = tid >> 5;
  const int hrow0 = prefix[e] + base;        // global h2 row base for this block

  // ---- Phase 0: row metadata ----
  if (tid < ROWS) {
    int m = -1; float w = 0.f;
    if (base + tid < cnt) {
      int idx = list[e * CAP + base + tid];
      m = idx / TOPK;
      w = topw[idx];
    }
    rowM[tid] = m; rowW[tid] = w;
  }
  __syncthreads();

  // ---- Phase 1: async-gather A tile (ROWS x K_DIM f16) into LDS ----
  {
    const unsigned ldsA = (unsigned)(uintptr_t)Atile;
    for (int it = 0; it < (ROWS * K_DIM) / (256 * 8); ++it) {   // 64 iters, 16B each
      int flat = (it * 256 + tid) * 8;
      int r = flat >> 10;
      int c = flat & (K_DIM - 1);
      int m = rowM[r]; if (m < 0) m = 0;                        // pad rows: dummy data, discarded later
      async_ld_b128(ldsA + r * (A_STR * 2) + c * 2,
                    hid + (long)m * K_DIM + c);
    }
    async_wait0();
  }
  __syncthreads();

  const f16* w1e = w1 + (long)e * N_DIM * K_DIM;
  const f16* w2e = w2 + (long)e * K_DIM * HALF_N;

  // ---- Phase 2: h1 = A @ w1^T ; h2 = silu(gate)*up -> workspace (via LDS staging) ----
  // wave: 16 gate cols + paired 16 up cols, all 128 rows; 8 waves * 16 = 128 cols/iter; 22 iters
  for (int itn = 0; itn < HALF_N / (8 * 16); ++itn) {
    const int ng = itn * 128 + wave * 16;
    v8f accG[8], accU[8];
    #pragma unroll
    for (int rg = 0; rg < 8; ++rg) { accG[rg] = {}; accU[rg] = {}; }

    for (int k = 0; k < K_DIM; k += 32) {
      v16h bg = load_b_glb(w1e, K_DIM, ng,          k);
      v16h bu = load_b_glb(w1e, K_DIM, ng + HALF_N, k);
      #pragma unroll
      for (int rg = 0; rg < 8; ++rg) {
        v16h a = load_a_lds(Atile, A_STR * 2, rg * 16, k);
        accG[rg] = wmma_f16(a, bg, accG[rg]);
        accU[rg] = wmma_f16(a, bu, accU[rg]);
      }
    }
    // silu(gate)*up -> LDS staging tile (row-major ROWS x 128)
    const int halfsel = lane >> 4;
    const int scol = wave * 16 + (lane & 15);
    #pragma unroll
    for (int rg = 0; rg < 8; ++rg) {
      #pragma unroll
      for (int i = 0; i < 8; ++i) {
        float h = silu_fast(accG[rg][i]) * accU[rg][i];
        int row = rg * 16 + halfsel * 8 + i;
        *(f16*)(Hstage + row * (HS_STR * 2) + scol * 2) = (f16)h;
      }
    }
    __syncthreads();
    // coalesced copy staging -> h2 workspace
    {
      int r = tid >> 1;
      int cbase = (tid & 1) * 64;
      f16* dst = h2ws + (long)(hrow0 + r) * H2_PAD + itn * 128 + cbase;
      const char* srow = Hstage + r * (HS_STR * 2) + cbase * 2;
      #pragma unroll
      for (int j = 0; j < 8; ++j)
        *(v8h*)(dst + j * 8) = *(const v8h*)(srow + j * 16);
    }
    __syncthreads();
  }
  __threadfence();   // make h2 workspace writes visible before phase-3 reads
  __syncthreads();

  // ---- Phase 3: out^T = w2 @ h2^T over padded H (3 chunks of 1024), atomic-add ----
  // wave: 32 output cols (2 x 16 kout rows of the transposed GEMM) x 128 token rows
  for (int ko = 0; ko < 4; ++ko) {
    const int kt0 = wave * 128 + ko * 32;
    v8f acc0[8], acc1[8];
    #pragma unroll
    for (int rg = 0; rg < 8; ++rg) { acc0[rg] = {}; acc1[rg] = {}; }

    for (int ch = 0; ch < 3; ++ch) {
      __syncthreads();                       // previous chunk fully consumed
      {
        const unsigned ldsC = (unsigned)(uintptr_t)Atile;
        const int c0 = ch * HC;
        for (int it = 0; it < (ROWS * HC) / (256 * 8); ++it) {  // 64 iters
          int flat = (it * 256 + tid) * 8;
          int r = flat >> 10;
          int c = flat & (HC - 1);
          async_ld_b128(ldsC + r * (C_STR * 2) + c * 2,
                        h2ws + (long)(hrow0 + r) * H2_PAD + c0 + c);
        }
        async_wait0();
      }
      __syncthreads();
      const int hg0 = ch * HC;
      for (int h = 0; h < HC; h += 32) {
        // A' = w2 rows (kout), K = h (contiguous). OOB h>=2816 multiplies zero h2 pad.
        v16h a0 = load_a_glb(w2e, HALF_N, kt0,      hg0 + h);
        v16h a1 = load_a_glb(w2e, HALF_N, kt0 + 16, hg0 + h);
        #pragma unroll
        for (int rg = 0; rg < 8; ++rg) {
          v16h b = load_b_lds(Atile, C_STR * 2, rg * 16, h);   // B' = h2 (token rows across lanes)
          acc0[rg] = wmma_f16(a0, b, acc0[rg]);
          acc1[rg] = wmma_f16(a1, b, acc1[rg]);
        }
      }
    }
    // epilogue: D'[kout][m]; each lane owns one token row (m fixed), 8 consecutive kout
    const int halfsel = lane >> 4;
    #pragma unroll
    for (int rg = 0; rg < 8; ++rg) {
      int mloc = rg * 16 + (lane & 15);
      int m = rowM[mloc];
      if (m >= 0) {
        float w = rowW[mloc];
        float* orow = out + (long)m * K_DIM;
        #pragma unroll
        for (int i = 0; i < 8; ++i) {
          atomicAdd(orow + kt0      + halfsel * 8 + i, acc0[rg][i] * w);
          atomicAdd(orow + kt0 + 16 + halfsel * 8 + i, acc1[rg][i] * w);
        }
      }
    }
  }
}

// ---------------- launcher ----------------
extern "C" void kernel_launch(void* const* d_in, const int* in_sizes, int n_in,
                              void* d_out, int out_size, void* d_ws, size_t ws_size,
                              hipStream_t stream) {
  const float* hid_f = (const float*)d_in[0];
  const float* w1_f  = (const float*)d_in[1];
  const float* w2_f  = (const float*)d_in[2];
  const float* topw  = (const float*)d_in[3];
  const int*   ids   = (const int*)d_in[4];
  float* out = (float*)d_out;

  // workspace layout
  char* ws = (char*)d_ws;
  int* counts = (int*)ws;                  // 8 ints
  int* prefix = (int*)(ws + 64);           // 8 ints
  int* list   = (int*)(ws + 256);          // E*CAP ints = 1 MiB
  size_t off  = 256 + (size_t)E_NUM * CAP * sizeof(int);
  f16* hid_h = (f16*)(ws + off); off += (size_t)M_TOK * K_DIM * sizeof(f16);
  f16* w1_h  = (f16*)(ws + off); off += (size_t)E_NUM * N_DIM * K_DIM * sizeof(f16);
  f16* w2_h  = (f16*)(ws + off); off += (size_t)E_NUM * K_DIM * HALF_N * sizeof(f16);
  f16* h2ws  = (f16*)(ws + off); off += (size_t)M_TOK * TOPK * H2_PAD * sizeof(f16);
  (void)off; (void)ws_size; (void)in_sizes; (void)n_in; (void)out_size;

  // 1) zero output, counters, and h2 workspace (zero pad columns feed phase-3 K-padding)
  zero_f32_kernel<<<8192, 256, 0, stream>>>(out, (long)M_TOK * K_DIM);
  zero_f32_kernel<<<1, 32, 0, stream>>>((float*)ws, 24);
  zero_f32_kernel<<<8192, 256, 0, stream>>>((float*)h2ws, (long)M_TOK * TOPK * H2_PAD / 2);

  // 2) fp32 -> f16 conversions
  cvt_f32_f16_kernel<<<4096, 256, 0, stream>>>(hid_f, hid_h, (long)M_TOK * K_DIM);
  cvt_f32_f16_kernel<<<4096, 256, 0, stream>>>(w1_f,  w1_h,  (long)E_NUM * N_DIM * K_DIM);
  cvt_f32_f16_kernel<<<4096, 256, 0, stream>>>(w2_f,  w2_h,  (long)E_NUM * K_DIM * HALF_N);

  // 3) routing
  route_scatter_kernel<<<(M_TOK * TOPK + 255) / 256, 256, 0, stream>>>(ids, counts, list);
  prefix_kernel<<<1, 32, 0, stream>>>(counts, prefix);

  // 4) fused expert GEMMs
  dim3 grid(CAP / ROWS, E_NUM);
  moe_fused_kernel<<<grid, 256, LDS_TOTAL, stream>>>(hid_h, w1_h, w2_h, topw,
                                                     counts, prefix, list, h2ws, out);
}
